// GINDefault_10737418240833
// MI455X (gfx1250) — matrix operand order repaired and checked by reading, not compiled, not run on hardware
//
#include <hip/hip_runtime.h>
#include <hip/hip_bf16.h>

#define NN     100000
#define NE     1600000
#define HIDC   128
#define NGRAPH 512
#define OUTC   10

#define MTILE  64
#define LPAD   8
#define LSTR   (HIDC + LPAD)   // 136 bf16 elements per padded LDS row

typedef __attribute__((ext_vector_type(16))) __bf16       v16bf;
typedef __attribute__((ext_vector_type(8)))  float        v8f;
typedef __attribute__((ext_vector_type(8)))  unsigned int v8u;

__device__ __forceinline__ __bf16 f2bf(float f) {
    unsigned u = __builtin_bit_cast(unsigned, f);
    unsigned r = (u + 0x7FFFu + ((u >> 16) & 1u)) >> 16;   // round-to-nearest-even
    return __builtin_bit_cast(__bf16, (unsigned short)r);
}

// ---------------------------------------------------------------- copy (agg = h), layer 0 only
__global__ void gin_copy_f4(float4* __restrict__ dst, const float4* __restrict__ src, int n4) {
    int i = blockIdx.x * blockDim.x + threadIdx.x;
    if (i < n4) dst[i] = src[i];
}

// ---------------------------------------------------------------- edge scatter-add
// one wave per edge: lane handles 4 channels (float4 gather + 4 float atomics, L2-resident)
__global__ void gin_scatter(float* __restrict__ agg, const float* __restrict__ h,
                            const int* __restrict__ srcI, const int* __restrict__ dstI,
                            int n_edges) {
    long long g = (long long)blockIdx.x * blockDim.x + threadIdx.x;
    int e    = (int)(g >> 5);
    int lane = (int)(g & 31);
    if (e >= n_edges) return;
    int s = srcI[e], d = dstI[e];
    float4 v = reinterpret_cast<const float4*>(h + (size_t)s * HIDC)[lane];
    float* o = agg + (size_t)d * HIDC + (size_t)lane * 4;
    atomicAdd(o + 0, v.x); atomicAdd(o + 1, v.y);
    atomicAdd(o + 2, v.z); atomicAdd(o + 3, v.w);
}

// ---------------------------------------------------------------- fused MLP (WMMA bf16)
// z (in-place): h_out = [relu]( relu(z@W1+b1) @ W2 + b2 );  optional dup-write to out2
// Weight B-fragments are loop-invariant -> kept in 64 VGPRs; LDS holds only activation tiles.
__global__ __launch_bounds__(256)
void gin_mlp_wmma(float* __restrict__ z, float* __restrict__ out2,
                  const float* __restrict__ w1, const float* __restrict__ b1,
                  const float* __restrict__ w2, const float* __restrict__ b2,
                  int n_nodes, int relu_out) {
    __shared__ __bf16 sIn [MTILE * LSTR];
    __shared__ __bf16 sMid[MTILE * LSTR];

    const int wave = threadIdx.x >> 5;   // 0..7 -> owns 16 output columns
    const int lane = threadIdx.x & 31;
    const int half = lane >> 4;          // ISA 16-bit fragment lane group
    const int ln   = lane & 15;
    const int ncol = wave * 16 + ln;

    // ---- one-time: weight B-fragments straight from global into registers ----
    // B (32x16 KxN) layout: lane(half,ln), VGPR j holds B[kk+half*16+2j][ncol], B[..+1][ncol]
    v16bf bw1[4], bw2[4];
    #pragma unroll
    for (int t = 0; t < 4; ++t) {
        v8u u1, u2;
        #pragma unroll
        for (int j = 0; j < 8; ++j) {
            int k = t * 32 + (half << 4) + (j << 1);
            unsigned short lo1 = __builtin_bit_cast(unsigned short, f2bf(w1[(size_t)k * HIDC + ncol]));
            unsigned short hi1 = __builtin_bit_cast(unsigned short, f2bf(w1[(size_t)(k + 1) * HIDC + ncol]));
            u1[j] = ((unsigned)hi1 << 16) | lo1;
            unsigned short lo2 = __builtin_bit_cast(unsigned short, f2bf(w2[(size_t)k * HIDC + ncol]));
            unsigned short hi2 = __builtin_bit_cast(unsigned short, f2bf(w2[(size_t)(k + 1) * HIDC + ncol]));
            u2[j] = ((unsigned)hi2 << 16) | lo2;
        }
        bw1[t] = __builtin_bit_cast(v16bf, u1);
        bw2[t] = __builtin_bit_cast(v16bf, u2);
    }
    const float bias1 = b1[ncol];
    const float bias2 = b2[ncol];

    for (int m0 = blockIdx.x * MTILE; m0 < n_nodes; m0 += (int)gridDim.x * MTILE) {
        __syncthreads();
        // stage z tile -> bf16 LDS (zero-pad tail rows)
        for (int i = threadIdx.x; i < MTILE * (HIDC / 4); i += 256) {
            int r  = i >> 5;           // HIDC/4 == 32
            int c4 = i & 31;
            __bf16* p = &sIn[r * LSTR + c4 * 4];
            if (m0 + r < n_nodes) {
                float4 v = reinterpret_cast<const float4*>(z + (size_t)(m0 + r) * HIDC)[c4];
                p[0] = f2bf(v.x); p[1] = f2bf(v.y); p[2] = f2bf(v.z); p[3] = f2bf(v.w);
            } else {
                __bf16 zb = __builtin_bit_cast(__bf16, (unsigned short)0);
                p[0] = zb; p[1] = zb; p[2] = zb; p[3] = zb;
            }
        }
        __syncthreads();

        // ---- GEMM1: mid = relu(z @ W1 + b1) ----
        #pragma unroll
        for (int ms = 0; ms < MTILE / 16; ++ms) {
            v8f acc = {};
            const int arow = ms * 16 + ln;
            #pragma unroll
            for (int t = 0; t < 4; ++t) {
                v8u au;
                #pragma unroll
                for (int j = 0; j < 8; ++j) {
                    int ka = t * 32 + (half << 4) + (j << 1);
                    au[j] = *reinterpret_cast<const unsigned int*>(&sIn[arow * LSTR + ka]);
                }
                acc = __builtin_amdgcn_wmma_f32_16x16x32_bf16(false, __builtin_bit_cast(v16bf, au),
                                                              false, bw1[t], (short)0, acc,
                                                              false, false);
            }
            #pragma unroll
            for (int r = 0; r < 8; ++r) {
                int mrow = ms * 16 + half * 8 + r;    // D layout: VGPR r -> M = r + 8*half
                float v = acc[r] + bias1;
                v = v > 0.0f ? v : 0.0f;
                sMid[mrow * LSTR + ncol] = f2bf(v);
            }
        }
        __syncthreads();

        // ---- GEMM2: out = mid @ W2 + b2 (optional inter-layer relu) ----
        #pragma unroll
        for (int ms = 0; ms < MTILE / 16; ++ms) {
            v8f acc = {};
            const int arow = ms * 16 + ln;
            #pragma unroll
            for (int t = 0; t < 4; ++t) {
                v8u au;
                #pragma unroll
                for (int j = 0; j < 8; ++j) {
                    int ka = t * 32 + (half << 4) + (j << 1);
                    au[j] = *reinterpret_cast<const unsigned int*>(&sMid[arow * LSTR + ka]);
                }
                acc = __builtin_amdgcn_wmma_f32_16x16x32_bf16(false, __builtin_bit_cast(v16bf, au),
                                                              false, bw2[t], (short)0, acc,
                                                              false, false);
            }
            #pragma unroll
            for (int r = 0; r < 8; ++r) {
                int mrow = ms * 16 + half * 8 + r;
                float v = acc[r] + bias2;
                if (relu_out) v = v > 0.0f ? v : 0.0f;
                if (m0 + mrow < n_nodes) {
                    size_t idx = (size_t)(m0 + mrow) * HIDC + ncol;
                    z[idx] = v;
                    if (out2) out2[idx] = v;   // next layer's agg init (saves a copy pass)
                }
            }
        }
    }
}

// ---------------------------------------------------------------- pooling
__global__ void gin_zero(float* __restrict__ p, int n) {
    int i = blockIdx.x * blockDim.x + threadIdx.x;
    if (i < n) p[i] = 0.0f;
}

__global__ void gin_pool(const float* __restrict__ h, const int* __restrict__ batch,
                         float* __restrict__ pooled, int n_nodes) {
    long long g = (long long)blockIdx.x * blockDim.x + threadIdx.x;
    int node = (int)(g >> 5);
    int lane = (int)(g & 31);
    if (node >= n_nodes) return;
    int b = batch[node];
    float4 v = reinterpret_cast<const float4*>(h + (size_t)node * HIDC)[lane];
    float* o = pooled + (size_t)b * HIDC + (size_t)lane * 4;
    atomicAdd(o + 0, v.x); atomicAdd(o + 1, v.y);
    atomicAdd(o + 2, v.z); atomicAdd(o + 3, v.w);
}

// ---------------------------------------------------------------- final linear [512x128]@[128x10]
__global__ void gin_final(const float* __restrict__ pooled, const float* __restrict__ lw,
                          const float* __restrict__ lb, float* __restrict__ out) {
    int g = blockIdx.x * blockDim.x + threadIdx.x;
    if (g >= NGRAPH * OUTC) return;
    int r = g / OUTC, c = g % OUTC;
    float acc = lb[c];
    #pragma unroll 8
    for (int k = 0; k < HIDC; ++k)
        acc += pooled[r * HIDC + k] * lw[k * OUTC + c];
    out[g] = acc;
}

// ---------------------------------------------------------------- launcher
extern "C" void kernel_launch(void* const* d_in, const int* in_sizes, int n_in,
                              void* d_out, int out_size, void* d_ws, size_t ws_size,
                              hipStream_t stream) {
    const float* x     = (const float*)d_in[0];
    const int*   ei    = (const int*)d_in[1];    // [2, NE] (int32 on device)
    const int*   batch = (const int*)d_in[2];
    const float* w1[3] = {(const float*)d_in[3],  (const float*)d_in[7],  (const float*)d_in[11]};
    const float* b1[3] = {(const float*)d_in[4],  (const float*)d_in[8],  (const float*)d_in[12]};
    const float* w2[3] = {(const float*)d_in[5],  (const float*)d_in[9],  (const float*)d_in[13]};
    const float* b2[3] = {(const float*)d_in[6],  (const float*)d_in[10], (const float*)d_in[14]};
    const float* lw    = (const float*)d_in[15];
    const float* lb    = (const float*)d_in[16];

    const int* srcI = ei;
    const int* dstI = ei + NE;

    const size_t featBytes = (size_t)NN * HIDC * sizeof(float);
    float* bufA   = (float*)d_ws;
    float* bufB   = (float*)((char*)d_ws + featBytes);
    float* pooled = (float*)((char*)d_ws + 2 * featBytes);

    dim3 blk(256);
    const int n4       = NN * HIDC / 4;
    const int copyGrid = (n4 + 255) / 256;
    const int scatGrid = (int)(((long long)NE * 32 + 255) / 256);
    const int mlpGrid  = (NN + MTILE - 1) / MTILE;

    // Layer 0: agg(A) = x + scatter(x); h1 = MLP(A), dup into B (= layer-1 agg init)
    gin_copy_f4<<<copyGrid, blk, 0, stream>>>((float4*)bufA, (const float4*)x, n4);
    gin_scatter<<<scatGrid, blk, 0, stream>>>(bufA, x, srcI, dstI, NE);
    gin_mlp_wmma<<<mlpGrid, blk, 0, stream>>>(bufA, bufB, w1[0], b1[0], w2[0], b2[0], NN, 1);

    // Layer 1: agg(B) += scatter(h1=A); h2 = MLP(B), dup into A (= layer-2 agg init)
    gin_scatter<<<scatGrid, blk, 0, stream>>>(bufB, bufA, srcI, dstI, NE);
    gin_mlp_wmma<<<mlpGrid, blk, 0, stream>>>(bufB, bufA, w1[1], b1[1], w2[1], b2[1], NN, 1);

    // Layer 2: agg(A) += scatter(h2=B); h3 = MLP(A), no dup, no final relu
    gin_scatter<<<scatGrid, blk, 0, stream>>>(bufA, bufB, srcI, dstI, NE);
    gin_mlp_wmma<<<mlpGrid, blk, 0, stream>>>(bufA, nullptr, w1[2], b1[2], w2[2], b2[2], NN, 0);

    gin_zero<<<(NGRAPH * HIDC + 255) / 256, blk, 0, stream>>>(pooled, NGRAPH * HIDC);
    gin_pool<<<(int)(((long long)NN * 32 + 255) / 256), blk, 0, stream>>>(bufA, batch, pooled, NN);
    gin_final<<<(NGRAPH * OUTC + 255) / 256, blk, 0, stream>>>(pooled, lw, lb, (float*)d_out);
}
